// PointNet2PartSeg_7516192768036
// MI455X (gfx1250) — matrix operand order repaired and checked by reading, not compiled, not run on hardware
//
#include <hip/hip_runtime.h>
#include <hip/hip_bf16.h>

// PointNet++ MSG part-seg forward for gfx1250 (MI455X, wave32).
// All MLPs -> v_wmma_f32_16x16x32_f16. Weights pre-packed to zero-padded f16
// (BN folded into per-channel scale/bias); activations kept as zero-padded f16
// so WMMA fragment loads are plain aligned vector loads (no guards/converts).
// Each wave computes TWO 16-row M-tiles x four 16-col N-tiles: B fragments are
// loaded once per k-chunk and reused by both M-tiles (8 WMMAs per chunk).

typedef __attribute__((ext_vector_type(16))) _Float16 v16h;
typedef __attribute__((ext_vector_type(8)))  _Float16 v8h;
typedef __attribute__((ext_vector_type(8)))  float    v8f;

#define BATCH 8
#define NPTS  2048
#define CHUNK 16384

static __host__ __device__ inline int pad32i(int x) { return (x + 31) & ~31; }

// ---------------------------------------------------------------------------
// Pack one layer: Wh[n*Kp+k] = f16(W[n][k]) (zeros outside), and fold BN:
//   scale = gamma*rsqrt(var+eps), bias = (b-mean)*scale+beta   (bn=1)
//   scale = 1, bias = b                                        (bn=0)
// Pad channels get scale=1,bias=0 -> pad outputs are exactly 0.
// ---------------------------------------------------------------------------
__global__ void pack_layer(const float* __restrict__ W, const float* __restrict__ b,
                           const float* __restrict__ beta, const float* __restrict__ gamma,
                           const float* __restrict__ mean, const float* __restrict__ var,
                           int Cin, int Cout, int Kp, int Np, int bn,
                           _Float16* __restrict__ Wh,
                           float* __restrict__ sc, float* __restrict__ bs)
{
  long long tid = (long long)blockIdx.x * blockDim.x + threadIdx.x;
  long long total = (long long)Np * Kp;
  if (tid >= total) return;
  int k = (int)(tid % Kp);
  int n = (int)(tid / Kp);
  _Float16 v = (_Float16)0.0f;
  if (n < Cout && k < Cin) v = (_Float16)W[(long long)n * Cin + k];
  Wh[tid] = v;
  if (k == 0) {
    float s = 1.0f, bb = 0.0f;
    if (n < Cout) {
      if (bn) { s = gamma[n] * rsqrtf(var[n] + 1e-5f); bb = (b[n] - mean[n]) * s + beta[n]; }
      else    { bb = b[n]; }
    }
    sc[n] = s; bs[n] = bb;
  }
}

// ---------------------------------------------------------------------------
// Y = act((X * W^T) * scale + bias). X: M x Kp f16 (zero-padded), Wh: Np x Kp
// f16 (zero-padded). 128-thread block = 4 waves; wave = two 16-row M-tiles x
// four 16-col N-tiles (block covers 128 rows x 64 cols).
// ---------------------------------------------------------------------------
__global__ __launch_bounds__(128) void wmma_linear(
    const _Float16* __restrict__ X,
    const _Float16* __restrict__ Wh, int Kp,
    const float* __restrict__ sc, const float* __restrict__ bs,
    void* __restrict__ Yv, int ldY,
    int M, int padTo, int relu, int outf32)
{
  const int lane  = threadIdx.x;                 // 0..31
  const int wave  = threadIdx.y;                 // 0..3
  const int m0    = (blockIdx.x * 8 + wave * 2) * 16;   // first M-tile
  if (m0 >= M) return;                           // uniform per wave
  const int n0    = blockIdx.y * 64;
  const int lhalf = lane >> 4;
  const int l15   = lane & 15;
  int rowA = m0 + l15;       if (rowA >= M) rowA = M - 1;   // clamped; discarded later
  int rowB = m0 + 16 + l15;  if (rowB >= M) rowB = M - 1;
  const _Float16* xrowA = X  + (size_t)rowA * Kp;
  const _Float16* xrowB = X  + (size_t)rowB * Kp;
  const _Float16* wbase = Wh + (size_t)(n0 + l15) * Kp;

  v8f acc[8] = {};
  for (int k0 = 0; k0 < Kp; k0 += 32) {
    if (k0 + 32 < Kp) {
      __builtin_prefetch(xrowA + k0 + 32, 0, 0);
      __builtin_prefetch(xrowB + k0 + 32, 0, 0);
    }
    // A fragments: lane half picks 16 contiguous K (k = 16*lhalf + s), matching B
    v16h aA = *(const v16h*)(xrowA + k0 + lhalf * 16);
    v16h aB = *(const v16h*)(xrowB + k0 + lhalf * 16);
    // B fragments: 16 contiguous halfs of W rows (cols n0+t*16+l15)
    v16h bf[4];
#pragma unroll
    for (int t = 0; t < 4; ++t)
      bf[t] = *(const v16h*)(wbase + (size_t)t * 16 * Kp + k0 + lhalf * 16);
    // 8 independent WMMAs reuse the 6 fragments just loaded
#pragma unroll
    for (int t = 0; t < 4; ++t) {
      acc[t] = __builtin_amdgcn_wmma_f32_16x16x32_f16(
          false, aA, false, bf[t], (short)0, acc[t], false, false);
      acc[t + 4] = __builtin_amdgcn_wmma_f32_16x16x32_f16(
          false, aB, false, bf[t], (short)0, acc[t + 4], false, false);
    }
  }

  float*    Yf = (float*)Yv;
  _Float16* Yh = (_Float16*)Yv;
#pragma unroll
  for (int mt = 0; mt < 2; ++mt) {
    int mb = m0 + mt * 16;
#pragma unroll
    for (int t = 0; t < 4; ++t) {
      int ncol = n0 + t * 16 + l15;
      if (ncol >= padTo) continue;
      float s = sc[ncol], b = bs[ncol];
#pragma unroll
      for (int i = 0; i < 8; ++i) {
        int row = mb + i + (lhalf ? 8 : 0);
        if (row < M) {
          float v = acc[mt * 4 + t][i] * s + b;
          if (relu) v = fmaxf(v, 0.0f);
          if (outf32) Yf[(size_t)row * ldY + ncol] = v;
          else        Yh[(size_t)row * ldY + ncol] = (_Float16)v;
        }
      }
    }
  }
}

// ---------------------------------------------------------------------------
// xyz (B,6,N) -> l0_xyz (B,N,3), l0_points (B,N,6)   (f32)
// ---------------------------------------------------------------------------
__global__ void prep_transpose(const float* __restrict__ xyz,
                               float* __restrict__ l0x, float* __restrict__ l0p)
{
  long long tid = (long long)blockIdx.x * blockDim.x + threadIdx.x;
  long long total = (long long)BATCH * NPTS * 6;
  if (tid >= total) return;
  int c = (int)(tid % 6);
  int n = (int)((tid / 6) % NPTS);
  int b = (int)(tid / (6LL * NPTS));
  float v = xyz[((long long)b * 6 + c) * NPTS + n];
  l0p[((long long)b * NPTS + n) * 6 + c] = v;
  if (c < 3) l0x[((long long)b * NPTS + n) * 3 + c] = v;
}

// ---------------------------------------------------------------------------
// Farthest point sampling (jnp.argmax tie rule: lowest index; first pick = 0)
// ---------------------------------------------------------------------------
__global__ __launch_bounds__(256) void fps_kernel(
    const float* __restrict__ xyz, int Np, int npoint, int* __restrict__ idx)
{
  __shared__ float dist[2048];
  __shared__ float rval[256];
  __shared__ int   ridx[256];
  int b = blockIdx.x, t = threadIdx.x;
  const float* base = xyz + (long long)b * Np * 3;
  for (int i = t; i < Np; i += 256) dist[i] = 1e10f;
  __syncthreads();
  int sel = 0;
  for (int it = 0; it < npoint; ++it) {
    if (t == 0) idx[b * npoint + it] = sel;
    float cx = base[sel * 3 + 0], cy = base[sel * 3 + 1], cz = base[sel * 3 + 2];
    float bestv = -1.0f; int besti = 0x7fffffff;
    for (int i = t; i < Np; i += 256) {
      float dx = base[i * 3 + 0] - cx;
      float dy = base[i * 3 + 1] - cy;
      float dz = base[i * 3 + 2] - cz;
      float nd = fminf(dist[i], dx * dx + dy * dy + dz * dz);
      dist[i] = nd;
      if (nd > bestv) { bestv = nd; besti = i; }
    }
    rval[t] = bestv; ridx[t] = besti;
    __syncthreads();
    for (int s = 128; s > 0; s >>= 1) {
      if (t < s) {
        if (rval[t + s] > rval[t] ||
            (rval[t + s] == rval[t] && ridx[t + s] < ridx[t])) {
          rval[t] = rval[t + s]; ridx[t] = ridx[t + s];
        }
      }
      __syncthreads();
    }
    sel = ridx[0];
    __syncthreads();
  }
}

__global__ void gather3(const float* __restrict__ xyz, const int* __restrict__ idx,
                        int Np, int S, float* __restrict__ out)
{
  long long tid = (long long)blockIdx.x * blockDim.x + threadIdx.x;
  long long total = (long long)BATCH * S * 3;
  if (tid >= total) return;
  int c = (int)(tid % 3);
  long long g = tid / 3;
  int b = (int)(g / S);
  int i = idx[g];
  out[g * 3 + c] = xyz[((long long)b * Np + i) * 3 + c];
}

// ---------------------------------------------------------------------------
// Ball query: one wave per center; first K in-radius indices ascending,
// padded with the first hit.
// ---------------------------------------------------------------------------
__global__ __launch_bounds__(256) void ball_query(
    const float* __restrict__ xyz, const float* __restrict__ nxyz,
    int S, int Np, float r2, int K, int* __restrict__ gi, int Mctr)
{
  int lane = threadIdx.x;
  int cw = blockIdx.x * blockDim.y + threadIdx.y;
  if (cw >= Mctr) return;
  int b = cw / S;
  float cx = nxyz[cw * 3 + 0], cy = nxyz[cw * 3 + 1], cz = nxyz[cw * 3 + 2];
  const float* base = xyz + (long long)b * Np * 3;
  int* out = gi + (long long)cw * K;
  int cnt = 0, first = -1;
  for (int bi = 0; bi < Np && cnt < K; bi += 32) {
    int i = bi + lane;
    bool pred = false;
    if (i < Np) {
      float dx = base[i * 3 + 0] - cx;
      float dy = base[i * 3 + 1] - cy;
      float dz = base[i * 3 + 2] - cz;
      pred = (dx * dx + dy * dy + dz * dz) <= r2;
    }
    unsigned mask = (unsigned)__ballot(pred);
    if (first < 0 && mask) first = bi + __ffs((int)mask) - 1;
    if (pred) {
      int pos = cnt + __popc(mask & ((1u << lane) - 1u));
      if (pos < K) out[pos] = i;
    }
    cnt += __popc(mask);
  }
  if (cnt > K) cnt = K;
  int f = (first < 0) ? 0 : first;
  for (int p = cnt + lane; p < K; p += 32) out[p] = f;
}

// ---------------------------------------------------------------------------
// Grouped features chunk -> f16 rows: [points(C), xyz-center(3), zeros(pad)]
// ---------------------------------------------------------------------------
__global__ void group_features(
    const float* __restrict__ pts, int ldP, int C,
    const float* __restrict__ xyz, const float* __restrict__ nxyz,
    const int* __restrict__ gi, int S, int K, int Np,
    long long cs, int cp, _Float16* __restrict__ out, int ldX)
{
  long long tid = (long long)blockIdx.x * blockDim.x + threadIdx.x;
  long long total = (long long)cp * ldX;
  if (tid >= total) return;
  int c = (int)(tid % ldX);
  long long r = tid / ldX;
  long long gidx = cs + r;
  long long g = gidx / K;
  int j = (int)(gidx % K);
  int b = (int)(g / S);
  int i = gi[g * K + j];
  if (i < 0) i = 0;
  if (i >= Np) i = Np - 1;
  float v = 0.0f;
  if (c < C)          v = pts[((long long)b * Np + i) * ldP + c];
  else if (c < C + 3) v = xyz[((long long)b * Np + i) * 3 + (c - C)] - nxyz[g * 3 + (c - C)];
  out[r * ldX + c] = (_Float16)v;
}

// max over K consecutive f16 rows -> f32 out[(gstart+gl)*ldO + coff + c]
__global__ void maxpool_groups(
    const _Float16* __restrict__ in, int ldI, int K, int C,
    long long gstart, int ng, float* __restrict__ out, int ldO, int coff)
{
  long long tid = (long long)blockIdx.x * blockDim.x + threadIdx.x;
  long long total = (long long)ng * C;
  if (tid >= total) return;
  int c = (int)(tid % C);
  long long gl = tid / C;
  const _Float16* p = in + gl * K * (long long)ldI + c;
  float m = -3.0e38f;
  for (int j = 0; j < K; ++j) m = fmaxf(m, (float)p[(long long)j * ldI]);
  out[(gstart + gl) * ldO + coff + c] = m;
}

// f32 src cols -> f16 dst cols at column offset
__global__ void copy_cols(const float* __restrict__ src, int ldS,
                          _Float16* __restrict__ dst, int ldD, int dcoff,
                          long long Mtot, int C)
{
  long long tid = (long long)blockIdx.x * blockDim.x + threadIdx.x;
  long long total = Mtot * C;
  if (tid >= total) return;
  int c = (int)(tid % C);
  long long m = tid / C;
  dst[m * ldD + dcoff + c] = (_Float16)src[m * ldS + c];
}

__global__ void bcast_cols(const float* __restrict__ src, int ldS, int rowdiv,
                           _Float16* __restrict__ dst, int ldD, int dcoff,
                           long long Mtot, int C)
{
  long long tid = (long long)blockIdx.x * blockDim.x + threadIdx.x;
  long long total = Mtot * C;
  if (tid >= total) return;
  int c = (int)(tid % C);
  long long m = tid / C;
  dst[m * ldD + dcoff + c] = (_Float16)src[(m / rowdiv) * ldS + c];
}

// zero-fill pad columns [c0, ld) for M rows
__global__ void fillpad(_Float16* __restrict__ dst, int ld, int c0, long long Mtot)
{
  long long tid = (long long)blockIdx.x * blockDim.x + threadIdx.x;
  int pc = ld - c0;
  long long total = Mtot * pc;
  if (tid >= total) return;
  int c = (int)(tid % pc);
  long long m = tid / pc;
  dst[m * ld + c0 + c] = (_Float16)0.0f;
}

// 3-NN (squared dist, first-index tie rule) + normalized inverse-dist weights
__global__ void knn3(const float* __restrict__ q, const float* __restrict__ ref,
                     int Nq, int Nr, int Mtot,
                     int* __restrict__ idx3, float* __restrict__ w3)
{
  int m = blockIdx.x * blockDim.x + threadIdx.x;
  if (m >= Mtot) return;
  int b = m / Nq;
  float qx = q[m * 3 + 0], qy = q[m * 3 + 1], qz = q[m * 3 + 2];
  const float* rb = ref + (long long)b * Nr * 3;
  float d0 = 3.0e38f, d1 = 3.0e38f, d2 = 3.0e38f;
  int i0 = 0, i1 = 0, i2 = 0;
  for (int i = 0; i < Nr; ++i) {
    float dx = rb[i * 3 + 0] - qx;
    float dy = rb[i * 3 + 1] - qy;
    float dz = rb[i * 3 + 2] - qz;
    float d = dx * dx + dy * dy + dz * dz;
    if (d < d0)      { d2 = d1; i2 = i1; d1 = d0; i1 = i0; d0 = d; i0 = i; }
    else if (d < d1) { d2 = d1; i2 = i1; d1 = d;  i1 = i; }
    else if (d < d2) { d2 = d;  i2 = i; }
  }
  float w0 = 1.0f / (d0 + 1e-8f);
  float w1 = 1.0f / (d1 + 1e-8f);
  float w2 = 1.0f / (d2 + 1e-8f);
  float s = w0 + w1 + w2;
  idx3[m * 3 + 0] = i0; idx3[m * 3 + 1] = i1; idx3[m * 3 + 2] = i2;
  w3[m * 3 + 0] = w0 / s; w3[m * 3 + 1] = w1 / s; w3[m * 3 + 2] = w2 / s;
}

// dst[m*ldD+c] = sum_k w3[m][k] * feat[b][idx3[m][k]][c]   (f16 feat -> f16 dst)
__global__ void interp3(const _Float16* __restrict__ feat, int ldF, int Nr,
                        const int* __restrict__ idx3, const float* __restrict__ w3,
                        long long Mtot, int Nq, int C,
                        _Float16* __restrict__ dst, int ldD)
{
  long long tid = (long long)blockIdx.x * blockDim.x + threadIdx.x;
  long long total = Mtot * C;
  if (tid >= total) return;
  int c = (int)(tid % C);
  long long m = tid / C;
  int b = (int)(m / Nq);
  const _Float16* fb = feat + (long long)b * Nr * ldF;
  float v = w3[m * 3 + 0] * (float)fb[(long long)idx3[m * 3 + 0] * ldF + c]
          + w3[m * 3 + 1] * (float)fb[(long long)idx3[m * 3 + 1] * ldF + c]
          + w3[m * 3 + 2] * (float)fb[(long long)idx3[m * 3 + 2] * ldF + c];
  dst[m * ldD + c] = (_Float16)v;
}

__global__ void logsoftmax_rows(float* __restrict__ out, int Mtot, int C)
{
  int m = blockIdx.x * blockDim.x + threadIdx.x;
  if (m >= Mtot) return;
  float* p = out + (long long)m * C;
  float mx = -3.0e38f;
  for (int c = 0; c < C; ++c) mx = fmaxf(mx, p[c]);
  float s = 0.0f;
  for (int c = 0; c < C; ++c) s += __expf(p[c] - mx);
  float ls = __logf(s);
  for (int c = 0; c < C; ++c) p[c] = p[c] - mx - ls;
}

// ---------------------------------------------------------------------------
struct LayerDef { int base; int cin; int cout; int bn; };
static const LayerDef LDEFS[26] = {
  {2,153,128,1},{8,128,128,1},                        // 0,1  fp1
  {14,576,256,1},{20,256,128,1},                      // 2,3  fp2
  {26,1536,256,1},{32,256,256,1},                     // 4,5  fp3
  {38,128,128,1},                                     // 6    head1
  {44,128,50,0},                                      // 7    head2
  {46,9,32,1},{52,32,32,1},{58,32,64,1},              // 8-10  sa1 b0
  {64,9,64,1},{70,64,64,1},{76,64,128,1},             // 11-13 sa1 b1
  {82,9,64,1},{88,64,96,1},{94,96,128,1},             // 14-16 sa1 b2
  {100,323,128,1},{106,128,128,1},{112,128,256,1},    // 17-19 sa2 b0
  {118,323,128,1},{124,128,196,1},{130,196,256,1},    // 20-22 sa2 b1
  {136,515,256,1},{142,256,512,1},{148,512,1024,1}    // 23-25 sa3
};

extern "C" void kernel_launch(void* const* d_in, const int* in_sizes, int n_in,
                              void* d_out, int out_size, void* d_ws, size_t ws_size,
                              hipStream_t stream)
{
  (void)in_sizes; (void)out_size; (void)ws_size;
  const float* Pm[160];
  for (int i = 0; i < n_in && i < 160; ++i) Pm[i] = (const float*)d_in[i];
  const float* xyz_in = Pm[0];
  const float* cls    = Pm[1];

  // --- workspace carve ---
  char* wsb = (char*)d_ws;
  size_t off = 0;
  auto alloc = [&](size_t bytes) -> void* {
    off = (off + 255) & ~(size_t)255;
    void* p = wsb + off;
    off += bytes;
    return p;
  };
  float*     l0x    = (float*)    alloc((size_t)BATCH * NPTS * 3 * 4);
  float*     l0p    = (float*)    alloc((size_t)BATCH * NPTS * 6 * 4);
  int*       fpsidx = (int*)      alloc((size_t)BATCH * 512 * 4);
  float*     l1x    = (float*)    alloc((size_t)BATCH * 512 * 3 * 4);
  float*     l1p    = (float*)    alloc((size_t)BATCH * 512 * 320 * 4);
  float*     l2x    = (float*)    alloc((size_t)BATCH * 128 * 3 * 4);
  float*     l2p    = (float*)    alloc((size_t)BATCH * 128 * 512 * 4);
  float*     l3     = (float*)    alloc((size_t)8 * 1024 * 4);
  int*       gi     = (int*)      alloc((size_t)BATCH * 512 * 128 * 4);
  _Float16*  bufA   = (_Float16*) alloc((size_t)CHUNK * 352 * 2);
  _Float16*  bufB   = (_Float16*) alloc((size_t)CHUNK * 352 * 2);
  _Float16*  gbuf   = (_Float16*) alloc((size_t)1024 * 1024 * 2);
  _Float16*  sa3in  = (_Float16*) alloc((size_t)1024 * 544 * 2);
  _Float16*  fp3in  = (_Float16*) alloc((size_t)1024 * 1536 * 2);
  _Float16*  l2pn   = (_Float16*) alloc((size_t)1024 * 256 * 2);
  _Float16*  fp2in  = (_Float16*) alloc((size_t)4096 * 576 * 2);
  _Float16*  l1pn   = (_Float16*) alloc((size_t)4096 * 128 * 2);
  _Float16*  fp1in  = (_Float16*) alloc((size_t)16384 * 160 * 2);
  int*       idx3   = (int*)      alloc((size_t)16384 * 3 * 4);
  float*     w3     = (float*)    alloc((size_t)16384 * 3 * 4);

  // packed weights + folded BN per layer
  _Float16* Wh[26]; float* Sc[26]; float* Bs[26]; int KP[26], NP[26];
  for (int li = 0; li < 26; ++li) {
    const LayerDef& L = LDEFS[li];
    KP[li] = pad32i(L.cin);
    NP[li] = ((L.cout + 63) / 64) * 64;
    Wh[li] = (_Float16*)alloc((size_t)NP[li] * KP[li] * 2);
    Sc[li] = (float*)alloc((size_t)NP[li] * 4);
    Bs[li] = (float*)alloc((size_t)NP[li] * 4);
  }

  auto blocks = [](long long total) {
    return dim3((unsigned)((total + 255) / 256));
  };

  // pack all layers
  for (int li = 0; li < 26; ++li) {
    const LayerDef& L = LDEFS[li];
    const float* Wp = Pm[L.base];
    const float* bp = Pm[L.base + 1];
    const float* be = L.bn ? Pm[L.base + 2] : bp;
    const float* ga = L.bn ? Pm[L.base + 3] : bp;
    const float* me = L.bn ? Pm[L.base + 4] : bp;
    const float* va = L.bn ? Pm[L.base + 5] : bp;
    pack_layer<<<blocks((long long)NP[li] * KP[li]), 256, 0, stream>>>(
        Wp, bp, be, ga, me, va, L.cin, L.cout, KP[li], NP[li], L.bn,
        Wh[li], Sc[li], Bs[li]);
  }

  // gemm: X (M x KP[li] f16) -> Y; padTo = pad32(Cout) for f16 chains
  auto gemm = [&](int li, const _Float16* X, int M, void* Y, int ldY,
                  int padTo, int outf32) {
    dim3 g((M + 127) / 128, NP[li] / 64), blk(32, 4);
    wmma_linear<<<g, blk, 0, stream>>>(X, Wh[li], KP[li], Sc[li], Bs[li],
                                       Y, ldY, M, padTo, LDEFS[li].bn, outf32);
  };

  // SA branch: ball query -> chunked (group -> 3x wmma MLP -> maxpool)
  auto run_sa_branch = [&](const float* xyzl, int Np, const float* pts, int ldP,
                           int C, const float* nxyz, int S, float radius, int K,
                           int L0, int L1, int L2,
                           float* outp, int ldO, int coff) {
    int Mctr = BATCH * S;
    ball_query<<<dim3((Mctr + 7) / 8), dim3(32, 8), 0, stream>>>(
        xyzl, nxyz, S, Np, radius * radius, K, gi, Mctr);
    int ld0 = KP[L0], p0 = pad32i(LDEFS[L0].cout), p1 = pad32i(LDEFS[L1].cout);
    int c2 = LDEFS[L2].cout, p2 = pad32i(c2);
    long long Ptot = (long long)Mctr * K;
    for (long long cs = 0; cs < Ptot; cs += CHUNK) {
      int cp = (int)((Ptot - cs < CHUNK) ? (Ptot - cs) : CHUNK);
      group_features<<<blocks((long long)cp * ld0), 256, 0, stream>>>(
          pts, ldP, C, xyzl, nxyz, gi, S, K, Np, cs, cp, bufA, ld0);
      gemm(L0, bufA, cp, bufB, p0, p0, 0);
      gemm(L1, bufB, cp, bufA, p1, p1, 0);
      gemm(L2, bufA, cp, bufB, p2, p2, 0);
      int ng = cp / K;
      maxpool_groups<<<blocks((long long)ng * c2), 256, 0, stream>>>(
          bufB, p2, K, c2, cs / K, ng, outp, ldO, coff);
    }
  };

  // ---- stage 0 ----
  prep_transpose<<<blocks((long long)BATCH * NPTS * 6), 256, 0, stream>>>(
      xyz_in, l0x, l0p);

  // ---- sa1 ----
  fps_kernel<<<BATCH, 256, 0, stream>>>(l0x, NPTS, 512, fpsidx);
  gather3<<<blocks((long long)BATCH * 512 * 3), 256, 0, stream>>>(
      l0x, fpsidx, NPTS, 512, l1x);
  run_sa_branch(l0x, NPTS, l0p, 6, 6, l1x, 512, 0.1f, 32,  8,  9, 10, l1p, 320, 0);
  run_sa_branch(l0x, NPTS, l0p, 6, 6, l1x, 512, 0.2f, 64, 11, 12, 13, l1p, 320, 64);
  run_sa_branch(l0x, NPTS, l0p, 6, 6, l1x, 512, 0.4f, 128,14, 15, 16, l1p, 320, 192);

  // ---- sa2 ----
  fps_kernel<<<BATCH, 256, 0, stream>>>(l1x, 512, 128, fpsidx);
  gather3<<<blocks((long long)BATCH * 128 * 3), 256, 0, stream>>>(
      l1x, fpsidx, 512, 128, l2x);
  run_sa_branch(l1x, 512, l1p, 320, 320, l2x, 128, 0.4f, 64, 17, 18, 19, l2p, 512, 0);
  run_sa_branch(l1x, 512, l1p, 320, 320, l2x, 128, 0.8f, 128,20, 21, 22, l2p, 512, 256);

  // ---- sa3: concat(l2_xyz, l2_points) -> 256 -> 512 -> 1024 -> max over 128 ----
  copy_cols<<<blocks(1024LL * 3),   256, 0, stream>>>(l2x, 3,   sa3in, 544, 0, 1024, 3);
  copy_cols<<<blocks(1024LL * 512), 256, 0, stream>>>(l2p, 512, sa3in, 544, 3, 1024, 512);
  fillpad<<<blocks(1024LL * (544 - 515)), 256, 0, stream>>>(sa3in, 544, 515, 1024);
  gemm(23, sa3in, 1024, bufA, 256,  256,  0);
  gemm(24, bufA,  1024, bufB, 512,  512,  0);
  gemm(25, bufB,  1024, gbuf, 1024, 1024, 0);
  maxpool_groups<<<blocks(8LL * 1024), 256, 0, stream>>>(
      gbuf, 1024, 128, 1024, 0, 8, l3, 1024, 0);

  // ---- fp3: concat(l2_points, broadcast l3) -> 256 -> 256 ----
  copy_cols<<<blocks(1024LL * 512), 256, 0, stream>>>(l2p, 512, fp3in, 1536, 0, 1024, 512);
  bcast_cols<<<blocks(1024LL * 1024), 256, 0, stream>>>(l3, 1024, 128, fp3in, 1536, 512, 1024, 1024);
  gemm(4, fp3in, 1024, bufA, 256, 256, 0);
  gemm(5, bufA,  1024, l2pn, 256, 256, 0);

  // ---- fp2: 3-NN interp (l2 -> l1) + concat l1_points -> 256 -> 128 ----
  knn3<<<blocks(4096), 256, 0, stream>>>(l1x, l2x, 512, 128, 4096, idx3, w3);
  copy_cols<<<blocks(4096LL * 320), 256, 0, stream>>>(l1p, 320, fp2in, 576, 0, 4096, 320);
  interp3<<<blocks(4096LL * 256), 256, 0, stream>>>(
      l2pn, 256, 128, idx3, w3, 4096, 512, 256, fp2in + 320, 576);
  gemm(2, fp2in, 4096, bufA, 256, 256, 0);
  gemm(3, bufA,  4096, l1pn, 128, 128, 0);

  // ---- fp1: 3-NN interp (l1 -> l0) + concat(cls,xyz,points) -> 128 -> 128 ----
  knn3<<<blocks(16384), 256, 0, stream>>>(l0x, l1x, NPTS, 512, 16384, idx3, w3);
  bcast_cols<<<blocks(16384LL * 16), 256, 0, stream>>>(cls, 16, NPTS, fp1in, 160, 0, 16384, 16);
  copy_cols<<<blocks(16384LL * 3), 256, 0, stream>>>(l0x, 3, fp1in, 160, 16, 16384, 3);
  copy_cols<<<blocks(16384LL * 6), 256, 0, stream>>>(l0p, 6, fp1in, 160, 19, 16384, 6);
  interp3<<<blocks(16384LL * 128), 256, 0, stream>>>(
      l1pn, 128, 512, idx3, w3, 16384, NPTS, 128, fp1in + 25, 160);
  fillpad<<<blocks(16384LL * (160 - 153)), 256, 0, stream>>>(fp1in, 160, 153, 16384);
  gemm(0, fp1in, 16384, bufA, 128, 128, 0);
  gemm(1, bufA,  16384, bufB, 128, 128, 0);

  // ---- head: 128 -> 128 (BN+ReLU) -> 50 linear (f32) -> log_softmax ----
  gemm(6, bufB, 16384, bufA, 128, 128, 0);
  gemm(7, bufA, 16384, d_out, 50, 50, 1);
  logsoftmax_rows<<<blocks(16384), 256, 0, stream>>>((float*)d_out, 16384, 50);
}
